// MotifNetLearnBasis_51135880626179
// MI455X (gfx1250) — matrix operand order repaired and checked by reading, not compiled, not run on hardware
//
#include <hip/hip_runtime.h>
#include <hip/hip_bf16.h>
#include <stdint.h>

typedef __attribute__((ext_vector_type(16))) __bf16 v16bf;
typedef __attribute__((ext_vector_type(8)))  float  v8f;
typedef __attribute__((ext_vector_type(8)))  unsigned short ushort8;
typedef __attribute__((ext_vector_type(4)))  float  float4v;
typedef __attribute__((ext_vector_type(4)))  unsigned int v4u;
typedef __attribute__((ext_vector_type(8)))  int v8i;
typedef __attribute__((ext_vector_type(4)))  int v4i;

#if __has_builtin(__builtin_amdgcn_tensor_load_to_lds)
#define HAVE_TDM 1
#else
#define HAVE_TDM 0
#endif

static __device__ __forceinline__ unsigned short f2bf(float f) {
  unsigned u = __float_as_uint(f);
  unsigned r = u + 0x7FFFu + ((u >> 16) & 1u);
  return (unsigned short)(r >> 16);
}
static __device__ __forceinline__ float bf2f(unsigned short h) {
  return __uint_as_float(((unsigned)h) << 16);
}

union Frag { v16bf v; ushort8 u8[2]; };

#if HAVE_TDM
// Issue a TDM DMA of a contiguous blob (nbytes, multiple of 8) from global to
// LDS. D# per CDNA5 ISA ch.8: group0 = {count/type, lds_addr, global_addr},
// group1 = {data_size=8B, tensor_dim0=tile_dim0=nbytes/8, 1 row, stride}.
// Completion tracked via TENSORcnt (wait separately with s_wait_tensorcnt).
static __device__ __forceinline__ void tdm_issue_copy(const void* gsrc,
                                                      unsigned lds_addr,
                                                      unsigned nbytes) {
  unsigned long long ga = (unsigned long long)(uintptr_t)gsrc;
  unsigned n8 = nbytes >> 3;               // elements of 8 bytes
  v4u g0;
  g0.x = 1u;                               // count=1 (valid), no gather
  g0.y = lds_addr;                         // D#.lds_addr  (bits 63:32)
  g0.z = (unsigned)(ga & 0xffffffffull);   // global_addr[31:0]  (bits 95:64)
  g0.w = (unsigned)((ga >> 32) & 0x1ffffffull) | (2u << 30); // addr[56:32], type=2
  v8i g1;
  g1[0] = (int)(3u << 16);                           // data_size=3 -> 8 bytes
  g1[1] = (int)((n8 & 0xffffu) << 16);               // tensor_dim0[15:0] @ bits 63:48
  g1[2] = (int)(((n8 >> 16) & 0xffffu) | (1u << 16));// tensor_dim0[31:16], tensor_dim1=1
  g1[3] = (int)((n8 & 0xffffu) << 16);               // tile_dim0 @ bits 127:112
  g1[4] = 1;                                         // tile_dim1=1
  g1[5] = (int)n8;                                   // tensor_dim0_stride[31:0]
  g1[6] = 0;
  g1[7] = 0;
  v4i z4 = {0, 0, 0, 0};
#if defined(__clang_major__) && __clang_major__ >= 23
  v8i z8 = {0, 0, 0, 0, 0, 0, 0, 0};
  __builtin_amdgcn_tensor_load_to_lds(g0, g1, z4, z4, z8, 0);
#else
  __builtin_amdgcn_tensor_load_to_lds(g0, g1, z4, z4, 0);
#endif
}
#endif

// ---------------------------------------------------------------------------
// Pack f32 weights (CO,CI,3,3) -> bf16 [tap][cout][cin]  (B-fragment friendly)
// ---------------------------------------------------------------------------
__global__ void pack_w(const float* __restrict__ src,
                       unsigned short* __restrict__ dst, int CO, int CI) {
  int n = CO * CI * 9;
  int idx = blockIdx.x * blockDim.x + threadIdx.x;
  if (idx >= n) return;
  int tap = idx / (CO * CI);
  int rem = idx - tap * CO * CI;
  int co = rem / CI, ci = rem - co * CI;
  dst[idx] = f2bf(src[(co * CI + ci) * 9 + tap]);
}

// ---------------------------------------------------------------------------
// conv1: 1 -> 64 channels, direct (tiny K), writes NHWC bf16 + relu
// ---------------------------------------------------------------------------
__global__ __launch_bounds__(256) void conv1_k(const float* __restrict__ x,
                                               const float* __restrict__ w1,
                                               const float* __restrict__ b1,
                                               unsigned short* __restrict__ out) {
  __shared__ float sw[64 * 9];
  __shared__ float sb[64];
  int tid = threadIdx.x;
  for (int i = tid; i < 576; i += 256) sw[i] = w1[i];
  if (tid < 64) sb[tid] = b1[tid];
  __syncthreads();

  int pix = blockIdx.x * 256 + tid;
  int b = pix >> 14;
  int yx = pix & 16383;
  int y = yx >> 7, xx = yx & 127;
  float xin[9];
#pragma unroll
  for (int dy = 0; dy < 3; ++dy)
#pragma unroll
    for (int dx = 0; dx < 3; ++dx) {
      int gy = y + dy - 1, gx = xx + dx - 1;
      float v = 0.f;
      if (gy >= 0 && gy < 128 && gx >= 0 && gx < 128)
        v = x[(b * 128 + gy) * 128 + gx];
      xin[dy * 3 + dx] = v;
    }
#pragma unroll
  for (int c8 = 0; c8 < 8; ++c8) {
    ushort8 pk;
#pragma unroll
    for (int k = 0; k < 8; ++k) {
      int co = c8 * 8 + k;
      float acc = sb[co];
#pragma unroll
      for (int t = 0; t < 9; ++t) acc = fmaf(sw[co * 9 + t], xin[t], acc);
      pk[k] = f2bf(fmaxf(acc, 0.f));
    }
    *(ushort8*)&out[(size_t)pix * 64 + c8 * 8] = pk;
  }
}

// ---------------------------------------------------------------------------
// conv64: 64 -> 64, 3x3 SAME, implicit GEMM with V_WMMA_F32_16X16X32_BF16.
// WG = 8 waves, output tile 16 rows x 8 cols x 64 ch. 72 WMMA per wave.
// Weights (73.7 KB bf16) DMA'd into LDS by the Tensor Data Mover while the
// waves stage the zero-padded activation halo with vector loads.
// ---------------------------------------------------------------------------
__global__ __launch_bounds__(256) void conv64_k(
    const unsigned short* __restrict__ in,      // NHWC bf16
    const unsigned short* __restrict__ wp,      // [9][64][64] bf16
    const float* __restrict__ bias,
    const unsigned short* __restrict__ resid,   // NHWC bf16 (optional)
    int has_res,
    unsigned short* __restrict__ out) {
  __shared__ __align__(16) unsigned short sAct[18 * 10 * 64]; // halo tile
  __shared__ __align__(16) unsigned short sW[9 * 64 * 64];    // full weights

  const int tid = threadIdx.x;
  const int lane = tid & 31;
  const int wave = tid >> 5;
  const int b = blockIdx.z;
  const int tY = blockIdx.y;   // 8 tiles of 16 rows
  const int tX = blockIdx.x;   // 16 tiles of 8 cols

#if HAVE_TDM
  if (wave == 0)
    tdm_issue_copy(wp, (unsigned)(uintptr_t)&sW[0], 9 * 64 * 64 * 2);
#endif

  // stage activation halo tile 18x10x64 (zero-fill SAME padding)
  for (int i = tid; i < 18 * 10 * 8; i += 256) {
    int r = i / 80;
    int rem = i - r * 80;
    int c = rem >> 3;
    int k = rem & 7;
    int gy = tY * 16 + r - 1;
    int gx = tX * 8 + c - 1;
    ushort8 v = {0, 0, 0, 0, 0, 0, 0, 0};
    if (gy >= 0 && gy < 128 && gx >= 0 && gx < 128)
      v = *(const ushort8*)&in[(((size_t)(b * 128 + gy)) * 128 + gx) * 64 + k * 8];
    *(ushort8*)&sAct[(r * 10 + c) * 64 + k * 8] = v;
  }
#if HAVE_TDM
  if (wave == 0) __builtin_amdgcn_s_wait_tensorcnt(0);
#else
  for (int i = tid; i < (9 * 64 * 64) / 8; i += 256)
    *(ushort8*)&sW[i * 8] = *(const ushort8*)&wp[i * 8];
#endif
  __syncthreads();

  const int pg = wave >> 1;    // pixel group: rows 4*pg .. 4*pg+3
  const int cg = wave & 1;     // output-channel half (32 each)
  const int n16 = lane & 15;
  const int hi = lane >> 4;

  v8f acc00 = {}, acc01 = {}, acc10 = {}, acc11 = {};

  // A lane geometry (16-bit A 16x32 layout): row m = lane&15 within subtile
  const int ay0 = 4 * pg + (n16 >> 3);  // + 2*s for subtile s
  const int ax = n16 & 7;

#pragma unroll
  for (int tap = 0; tap < 9; ++tap) {
    const int dyi = tap / 3, dxi = tap % 3;
#pragma unroll
    for (int kc = 0; kc < 2; ++kc) {
      const int chA = kc * 32 + hi * 8;
      Frag a0, a1, w0, w1;
      {
        int base0 = ((ay0 + dyi) * 10 + (ax + dxi)) * 64;
        a0.u8[0] = *(const ushort8*)&sAct[base0 + chA];
        a0.u8[1] = *(const ushort8*)&sAct[base0 + chA + 16];
        int base1 = ((ay0 + 2 + dyi) * 10 + (ax + dxi)) * 64;
        a1.u8[0] = *(const ushort8*)&sAct[base1 + chA];
        a1.u8[1] = *(const ushort8*)&sAct[base1 + chA + 16];
      }
      {
        const int chB = kc * 32 + hi * 8;
        int wi0 = ((tap * 64) + cg * 32 + n16) * 64 + chB;
        w0.u8[0] = *(const ushort8*)&sW[wi0];
        w0.u8[1] = *(const ushort8*)&sW[wi0 + 16];
        int wi1 = ((tap * 64) + cg * 32 + 16 + n16) * 64 + chB;
        w1.u8[0] = *(const ushort8*)&sW[wi1];
        w1.u8[1] = *(const ushort8*)&sW[wi1 + 16];
      }
      acc00 = __builtin_amdgcn_wmma_f32_16x16x32_bf16(false, a0.v, false, w0.v, (short)0, acc00, false, false);
      acc01 = __builtin_amdgcn_wmma_f32_16x16x32_bf16(false, a0.v, false, w1.v, (short)0, acc01, false, false);
      acc10 = __builtin_amdgcn_wmma_f32_16x16x32_bf16(false, a1.v, false, w0.v, (short)0, acc10, false, false);
      acc11 = __builtin_amdgcn_wmma_f32_16x16x32_bf16(false, a1.v, false, w1.v, (short)0, acc11, false, false);
    }
  }

  // epilogue: bias (+residual) + relu, store bf16 NHWC
  const float bias0 = bias[cg * 32 + n16];
  const float bias1 = bias[cg * 32 + 16 + n16];
#pragma unroll
  for (int s = 0; s < 2; ++s) {
    v8f aJ0 = s ? acc10 : acc00;
    v8f aJ1 = s ? acc11 : acc01;
    const int gy = tY * 16 + 4 * pg + 2 * s + hi;
#pragma unroll
    for (int r = 0; r < 8; ++r) {
      const int gx = tX * 8 + r;
      size_t pixbase = ((size_t)((b * 128 + gy) * 128 + gx)) * 64;
      {
        size_t addr = pixbase + cg * 32 + n16;
        float f = aJ0[r] + bias0;
        if (has_res) f += bf2f(resid[addr]);
        out[addr] = f2bf(fmaxf(f, 0.f));
      }
      {
        size_t addr = pixbase + cg * 32 + 16 + n16;
        float f = aJ1[r] + bias1;
        if (has_res) f += bf2f(resid[addr]);
        out[addr] = f2bf(fmaxf(f, 0.f));
      }
    }
  }
}

// ---------------------------------------------------------------------------
// conv3 (64 -> 16) + channel softmax + multiply by x0, fused.
// WG tile 16x16 pixels; each wave 2 rows x 16 cols; 36 WMMA per wave.
// Softmax over the 16 lanes that hold one pixel's channels (wave32 halves).
// ---------------------------------------------------------------------------
__global__ __launch_bounds__(256) void conv3_k(
    const unsigned short* __restrict__ in,   // NHWC bf16
    const unsigned short* __restrict__ wp,   // [9][16][64] bf16
    const float* __restrict__ b3,
    const float* __restrict__ x0,            // (B,H,W) f32
    float* __restrict__ h16out) {            // (B,H,W,16) f32
  __shared__ __align__(16) unsigned short sAct[18 * 18 * 64];
  __shared__ __align__(16) unsigned short sW[9 * 16 * 64];
  const int tid = threadIdx.x, lane = tid & 31, wave = tid >> 5;
  const int b = blockIdx.z, tY = blockIdx.y, tX = blockIdx.x;

#if HAVE_TDM
  if (wave == 0)
    tdm_issue_copy(wp, (unsigned)(uintptr_t)&sW[0], 9 * 16 * 64 * 2);
#endif

  for (int i = tid; i < 18 * 18 * 8; i += 256) {
    int r = i / 144, rem = i - r * 144, c = rem >> 3, k = rem & 7;
    int gy = tY * 16 + r - 1, gx = tX * 16 + c - 1;
    ushort8 v = {0, 0, 0, 0, 0, 0, 0, 0};
    if (gy >= 0 && gy < 128 && gx >= 0 && gx < 128)
      v = *(const ushort8*)&in[(((size_t)(b * 128 + gy)) * 128 + gx) * 64 + k * 8];
    *(ushort8*)&sAct[(r * 18 + c) * 64 + k * 8] = v;
  }
#if HAVE_TDM
  if (wave == 0) __builtin_amdgcn_s_wait_tensorcnt(0);
#else
  for (int i = tid; i < (9 * 16 * 64) / 8; i += 256)
    *(ushort8*)&sW[i * 8] = *(const ushort8*)&wp[i * 8];
#endif
  __syncthreads();

  const int pg = wave;           // rows 2*pg, 2*pg+1
  const int n16 = lane & 15;
  const int hi = lane >> 4;

  v8f acc0 = {}, acc1 = {};
#pragma unroll
  for (int tap = 0; tap < 9; ++tap) {
    const int dyi = tap / 3, dxi = tap % 3;
#pragma unroll
    for (int kc = 0; kc < 2; ++kc) {
      const int chA = kc * 32 + hi * 8;
      Frag a0, a1, wf;
      {
        int base0 = ((2 * pg + dyi) * 18 + (n16 + dxi)) * 64;
        a0.u8[0] = *(const ushort8*)&sAct[base0 + chA];
        a0.u8[1] = *(const ushort8*)&sAct[base0 + chA + 16];
        int base1 = ((2 * pg + 1 + dyi) * 18 + (n16 + dxi)) * 64;
        a1.u8[0] = *(const ushort8*)&sAct[base1 + chA];
        a1.u8[1] = *(const ushort8*)&sAct[base1 + chA + 16];
      }
      {
        int wi = ((tap * 16) + n16) * 64 + kc * 32 + hi * 8;
        wf.u8[0] = *(const ushort8*)&sW[wi];
        wf.u8[1] = *(const ushort8*)&sW[wi + 16];
      }
      acc0 = __builtin_amdgcn_wmma_f32_16x16x32_bf16(false, a0.v, false, wf.v, (short)0, acc0, false, false);
      acc1 = __builtin_amdgcn_wmma_f32_16x16x32_bf16(false, a1.v, false, wf.v, (short)0, acc1, false, false);
    }
  }

  const float bv = b3[n16];
#pragma unroll
  for (int s = 0; s < 2; ++s) {
    v8f A = s ? acc1 : acc0;
    const int gy = tY * 16 + 2 * pg + s;
#pragma unroll
    for (int r = 0; r < 8; ++r) {
      const int gx = tX * 16 + r + 8 * hi;
      float v = A[r] + bv;
      float mx = v;
      for (int o = 1; o < 16; o <<= 1) mx = fmaxf(mx, __shfl_xor(mx, o, 16));
      float e = __expf(v - mx);
      float sum = e;
      for (int o = 1; o < 16; o <<= 1) sum += __shfl_xor(sum, o, 16);
      size_t pix = (size_t)(b * 128 + gy) * 128 + gx;
      h16out[pix * 16 + n16] = (e / sum) * x0[pix];
    }
  }
}

// ---------------------------------------------------------------------------
// basis expansion: out[b, y*4+p, x*4+q] = sum_c h16[b,y,x,c] * basisN[c,p,q]
// ---------------------------------------------------------------------------
__global__ __launch_bounds__(256) void einsum_k(const float* __restrict__ h16,
                                                const float* __restrict__ basis_w,
                                                float* __restrict__ out) {
  __shared__ float sB[256];
  __shared__ float sSum[16];
  int tid = threadIdx.x;
  sB[tid] = __expf(basis_w[tid]);
  __syncthreads();
  if (tid < 16) {
    float s = 0.f;
    for (int q = 0; q < 16; ++q) s += sB[tid * 16 + q];
    sSum[tid] = s;
  }
  __syncthreads();
  sB[tid] = sB[tid] / sSum[tid >> 4];
  __syncthreads();

  int pix = blockIdx.x * 256 + tid;
  int b = pix >> 14;
  int yx = pix & 16383;
  int y = yx >> 7, x = yx & 127;

  float v[16];
  const float4v* hp = (const float4v*)&h16[(size_t)pix * 16];
#pragma unroll
  for (int i = 0; i < 4; ++i) {
    float4v t = hp[i];
    v[i * 4 + 0] = t[0]; v[i * 4 + 1] = t[1]; v[i * 4 + 2] = t[2]; v[i * 4 + 3] = t[3];
  }
#pragma unroll
  for (int p = 0; p < 4; ++p) {
    float4v o;
#pragma unroll
    for (int q = 0; q < 4; ++q) {
      float s = 0.f;
#pragma unroll
      for (int c = 0; c < 16; ++c) s = fmaf(v[c], sB[c * 16 + p * 4 + q], s);
      o[q] = s;
    }
    *(float4v*)&out[((size_t)(b * 512 + y * 4 + p)) * 512 + x * 4] = o;
  }
}

// ---------------------------------------------------------------------------
extern "C" void kernel_launch(void* const* d_in, const int* in_sizes, int n_in,
                              void* d_out, int out_size, void* d_ws, size_t ws_size,
                              hipStream_t stream) {
  const float* x   = (const float*)d_in[0];
  const float* w1  = (const float*)d_in[1];
  const float* b1  = (const float*)d_in[2];
  const float* w2  = (const float*)d_in[3];
  const float* b2  = (const float*)d_in[4];
  const float* rw1 = (const float*)d_in[5];
  const float* rb1 = (const float*)d_in[6];
  const float* rw2 = (const float*)d_in[7];
  const float* rb2 = (const float*)d_in[8];
  const float* w3  = (const float*)d_in[9];
  const float* b3  = (const float*)d_in[10];
  const float* bw  = (const float*)d_in[11];
  float* out = (float*)d_out;

  char* ws = (char*)d_ws;
  unsigned short* bufA = (unsigned short*)ws;                       // 33.5 MB bf16 NHWC
  unsigned short* bufB = (unsigned short*)(ws + (size_t)33554432);  // 33.5 MB
  float* h16          = (float*)(ws + (size_t)67108864);            // 16.8 MB
  unsigned short* wpk = (unsigned short*)(ws + (size_t)83886080);   // packed weights
  unsigned short* w2p = wpk;
  unsigned short* w3p = wpk + (size_t)36864 * 9;

  dim3 blk(256);
  pack_w<<<144, blk, 0, stream>>>(w2, w2p, 64, 64);
  for (int i = 0; i < 4; ++i) {
    pack_w<<<144, blk, 0, stream>>>(rw1 + (size_t)i * 36864, wpk + (size_t)36864 * (1 + i), 64, 64);
    pack_w<<<144, blk, 0, stream>>>(rw2 + (size_t)i * 36864, wpk + (size_t)36864 * (5 + i), 64, 64);
  }
  pack_w<<<36, blk, 0, stream>>>(w3, w3p, 16, 64);

  conv1_k<<<1024, blk, 0, stream>>>(x, w1, b1, bufA);

  dim3 g64(16, 8, 16);
  conv64_k<<<g64, blk, 0, stream>>>(bufA, w2p, b2, bufA, 0, bufB);
  for (int i = 0; i < 4; ++i) {
    conv64_k<<<g64, blk, 0, stream>>>(bufB, wpk + (size_t)36864 * (1 + i), rb1 + i * 64, bufB, 0, bufA);
    conv64_k<<<g64, blk, 0, stream>>>(bufA, wpk + (size_t)36864 * (5 + i), rb2 + i * 64, bufB, 1, bufB);
  }

  dim3 g3(8, 8, 16);
  conv3_k<<<g3, blk, 0, stream>>>(bufB, w3p, b3, x, h16);
  einsum_k<<<1024, blk, 0, stream>>>(h16, bw, out);
}
